// CRF_32598801776817
// MI455X (gfx1250) — matrix operand order repaired and checked by reading, not compiled, not run on hardware
//
#include <hip/hip_runtime.h>
#include <hip/hip_bf16.h>

// ---------------------------------------------------------------------------
// CRF NLL on gfx1250 (MI455X).
//   STATE=771 padded to 800. Forward scan done as scaled-prob GEMM with
//   v_wmma_f32_16x16x32_f16. 2 blocks x 16 batches (M=16 tiles, no waste).
//   kt outer (A fragment loaded once per kt, reused for 4 N-tiles), B tiles
//   prefetched one kt ahead with global_prefetch_b8, raw step output staged
//   in scaled-f16 LDS to keep VGPR pressure low (no spills).
// ---------------------------------------------------------------------------

#define S    771      // STATE
#define SP   800      // padded STATE (25 K-tiles of 32, 50 N-tiles of 16)
#define BV   32       // batch
#define TV   128      // time
#define NT   50       // N tiles (16 wide)
#define KT   25       // K tiles (32 deep)
#define NWAVES 16
#define BLOCK (NWAVES * 32)
#define EOS_IDX 1
#define SCALE_DN  0.0009765625f     // 2^-10, keeps raw sums inside f16 range
#define LOG_SCALE 6.93147180559945f // log(2^10)

typedef __attribute__((ext_vector_type(16))) _Float16 h16;
typedef __attribute__((ext_vector_type(8)))  _Float16 h8;
typedef __attribute__((ext_vector_type(8)))  float    f8;

// ---------------------------------------------------------------------------
// Prep: expT[i][j] = exp(trans[i][j]) stored in WMMA B-fragment order.
// Fragment (kt,nt): 512 halves; lane L owns halves [L*16 .. L*16+15]:
//   n = nt*16 + (L&15),  k = kt*32 + (L>>4)*16 + e   (e = 0..15)
// ---------------------------------------------------------------------------
__global__ void crf_prep_expT(const float* __restrict__ trans,
                              _Float16* __restrict__ expT) {
    int idx = blockIdx.x * blockDim.x + threadIdx.x;   // one per (tile,lane)
    if (idx >= KT * NT * 32) return;
    int lane = idx & 31;
    int tile = idx >> 5;
    int nt = tile % NT;
    int kt = tile / NT;
    int n = nt * 16 + (lane & 15);
    int kbase = kt * 32 + (lane >> 4) * 16;
    _Float16 tmp[16];
#pragma unroll
    for (int e = 0; e < 16; ++e) {
        int k = kbase + e;
        float v = 0.0f;
        if (k < S && n < S) v = __expf(trans[(size_t)k * S + n]);  // exp(-1e9)=0
        tmp[e] = (_Float16)v;
    }
    _Float16* dst = expT + (size_t)tile * 512 + lane * 16;
    *(h8*)dst       = *(const h8*)tmp;
    *(h8*)(dst + 8) = *(const h8*)(tmp + 8);
}

// ---------------------------------------------------------------------------
// Main scan kernel: one block = 16 batches (b0..b0+15), 16 waves.
// ---------------------------------------------------------------------------
__global__ __launch_bounds__(BLOCK) void crf_scan(
    const float* __restrict__ em, const int* __restrict__ tags,
    const float* __restrict__ mask, const float* __restrict__ trans,
    const _Float16* __restrict__ expT, float* __restrict__ partial) {

    __shared__ _Float16 sA[16 * SP];      // normalized probs (A operand), 25.6 KB
    __shared__ _Float16 sNew[16 * SP];    // raw step output, scaled f16, 25.6 KB
    __shared__ float Lrow[16];            // per-batch log scale
    __shared__ float scoreS[16];
    __shared__ float contrib[16];

    const int tid  = threadIdx.x;
    const int wid  = tid >> 5;
    const int lane = tid & 31;
    const int hi   = lane >> 4;           // which half-wave
    const int ln   = lane & 15;
    const int b0   = blockIdx.x * 16;

    // ---- Phase 0a: numerator score (wave w -> batch b0+w) ----
    {
        const int b = b0 + wid;
        const int* tb = tags + b * TV;
        const float* mb = mask + b * TV;
        float sc = 0.0f, msum = 0.0f;
        for (int t = lane; t < TV; t += 32) {
            float mk = mb[t];
            msum += mk;
            if (t >= 1) {
                int cur = tb[t], prev = tb[t - 1];
                sc += (em[((size_t)b * TV + t) * S + cur] +
                       trans[(size_t)prev * S + cur]) * mk;
            }
        }
        for (int off = 16; off >= 1; off >>= 1) {
            sc   += __shfl_xor(sc, off, 32);
            msum += __shfl_xor(msum, off, 32);
        }
        if (lane == 0) {
            int first = tb[0];
            sc += trans[first] + em[((size_t)b * TV) * S + first];   // BOS row 0
            int last_idx = (int)(msum + 0.5f) - 1;
            sc += trans[(size_t)tb[last_idx] * S + EOS_IDX];
            scoreS[wid] = sc;
        }
    }

    // ---- Phase 0b: t=0 init: s = exp(trans[BOS] + em[:,0] - max) ----
    {
        const int m = wid;
        const int b = b0 + m;
        for (int j = S + lane; j < SP; j += 32) sA[m * SP + j] = (_Float16)0.0f;
        float mx = -3.0e38f;
        for (int j = lane; j < S; j += 32)
            mx = fmaxf(mx, trans[j] + em[((size_t)b * TV) * S + j]);
        for (int off = 16; off >= 1; off >>= 1)
            mx = fmaxf(mx, __shfl_xor(mx, off, 32));
        for (int j = lane; j < S; j += 32) {
            float v = trans[j] + em[((size_t)b * TV) * S + j];
            sA[m * SP + j] = (_Float16)__expf(v - mx);
        }
        if (lane == 0) Lrow[m] = mx;
    }
    __syncthreads();

    // ---- Forward scan t = 1..T-1 ----
    for (int t = 1; t < TV; ++t) {
        // Phase A: GEMM. kt outer so each A fragment is loaded once and
        // reused for this wave's (up to) 4 N-tiles. acc[4] = 32 VGPRs.
        f8 acc[4];
#pragma unroll
        for (int ii = 0; ii < 4; ++ii)
            acc[ii] = (f8){0.f, 0.f, 0.f, 0.f, 0.f, 0.f, 0.f, 0.f};

#pragma unroll 1
        for (int kt = 0; kt < KT; ++kt) {
            // Prefetch next kt's B fragments (no VGPR cost, warms WGP$).
            if (kt + 1 < KT) {
#pragma unroll
                for (int ii = 0; ii < 4; ++ii) {
                    const int nt = wid + ii * NWAVES;
                    if (nt < NT)
                        __builtin_prefetch(
                            expT + (size_t)((kt + 1) * NT + nt) * 512 + lane * 16,
                            0, 3);
                }
            }
            // A fragment: row m=ln, K = kt*32 + hi*16 + {0..15}
            const _Float16* ap = &sA[ln * SP + kt * 32 + hi * 16];
            h8 alo = *(const h8*)ap;
            h8 ahi = *(const h8*)(ap + 8);
            h16 a = __builtin_shufflevector(alo, ahi,
                0,1,2,3,4,5,6,7,8,9,10,11,12,13,14,15);
#pragma unroll
            for (int ii = 0; ii < 4; ++ii) {
                const int nt = wid + ii * NWAVES;
                if (nt < NT) {
                    const _Float16* bp =
                        expT + (size_t)(kt * NT + nt) * 512 + lane * 16;
                    h8 blo = *(const h8*)bp;
                    h8 bhi = *(const h8*)(bp + 8);
                    h16 bfrag = __builtin_shufflevector(blo, bhi,
                        0,1,2,3,4,5,6,7,8,9,10,11,12,13,14,15);
                    acc[ii] = __builtin_amdgcn_wmma_f32_16x16x32_f16(
                        false, a, false, bfrag, (short)0, acc[ii], false, false);
                }
            }
        }
        // Epilogue: apply exp(emissions), store scaled raw values to LDS.
#pragma unroll
        for (int ii = 0; ii < 4; ++ii) {
            const int nt = wid + ii * NWAVES;
            if (nt < NT) {
                const int j = nt * 16 + ln;
#pragma unroll
                for (int r = 0; r < 8; ++r) {
                    const int m = r + hi * 8;
                    float v = 0.0f;
                    if (j < S)
                        v = acc[ii][r] *
                            __expf(em[((size_t)(b0 + m) * TV + t) * S + j]) *
                            SCALE_DN;
                    sNew[m * SP + j] = (_Float16)v;
                }
            }
        }
        __syncthreads();

        // Phase B: per-row renormalize. wave w -> batch row m=w.
        {
            const int m = wid;
            float mx = 0.0f;
            for (int j = lane; j < SP; j += 32)
                mx = fmaxf(mx, (float)sNew[m * SP + j]);
            for (int off = 16; off >= 1; off >>= 1)
                mx = fmaxf(mx, __shfl_xor(mx, off, 32));
            float mk = mask[(size_t)(b0 + m) * TV + t];
            if (mk > 0.0f) {                    // mask==0: keep old alphas
                if (lane == 0) Lrow[m] += __logf(mx) + LOG_SCALE;
                float inv = 1.0f / mx;
                for (int j = lane; j < SP; j += 32) {
                    float v = (float)sNew[m * SP + j] * inv;
                    sA[m * SP + j] = (_Float16)v;   // pads stay 0
                }
            }
        }
        __syncthreads();
    }

    // ---- log Z and block partial ----
    {
        const int m = wid;
        float sum = 0.0f;
        for (int j = lane; j < S; j += 32)
            sum += (float)sA[m * SP + j] * __expf(trans[(size_t)j * S + EOS_IDX]);
        for (int off = 16; off >= 1; off >>= 1) sum += __shfl_xor(sum, off, 32);
        if (lane == 0) {
            float logz = Lrow[m] + __logf(sum);
            contrib[m] = scoreS[m] - logz;
        }
    }
    __syncthreads();
    if (tid == 0) {
        float tot = 0.0f;
        for (int m2 = 0; m2 < 16; ++m2) tot += contrib[m2];
        partial[blockIdx.x] = tot;
    }
}

__global__ void crf_finish(const float* __restrict__ partial,
                           float* __restrict__ out) {
    out[0] = -(partial[0] + partial[1]);
}

// ---------------------------------------------------------------------------
extern "C" void kernel_launch(void* const* d_in, const int* in_sizes, int n_in,
                              void* d_out, int out_size, void* d_ws, size_t ws_size,
                              hipStream_t stream) {
    const float* em    = (const float*)d_in[0];   // (32,128,771) f32
    const int*   tags  = (const int*)d_in[1];     // (32,128) int
    const float* mask  = (const float*)d_in[2];   // (32,128) f32
    const float* trans = (const float*)d_in[3];   // (771,771) f32

    _Float16* expT = (_Float16*)d_ws;                         // 1,280,000 B
    float* partial = (float*)((char*)d_ws + (size_t)KT * NT * 512 * 2);

    const int prepThreads = KT * NT * 32;
    crf_prep_expT<<<(prepThreads + 255) / 256, 256, 0, stream>>>(trans, expT);
    crf_scan<<<2, BLOCK, 0, stream>>>(em, tags, mask, trans, expT, partial);
    crf_finish<<<1, 1, 0, stream>>>(partial, (float*)d_out);
}
